// SlidingAttention_35416300323573
// MI455X (gfx1250) — matrix operand order repaired and checked by reading, not compiled, not run on hardware
//
#include <hip/hip_runtime.h>

#define SEQ_LEN 4096
#define EMB     1024
#define KWIN    256
#define NWIN    240
#define NHEADS  16
#define DH      64

typedef __attribute__((ext_vector_type(16))) _Float16 v16h;
typedef __attribute__((ext_vector_type(8)))  _Float16 v8h;
typedef __attribute__((ext_vector_type(8)))  float    v8f;
typedef __attribute__((ext_vector_type(4)))  unsigned int u32x4;
typedef __attribute__((ext_vector_type(8)))  int i32x8;
typedef __attribute__((ext_vector_type(4)))  int i32x4;

#if __has_builtin(__builtin_amdgcn_tensor_load_to_lds)
#define HAVE_TDM 1
#endif

__device__ __forceinline__ v8f wmma32(v16h a, v16h b, v8f c) {
  return __builtin_amdgcn_wmma_f32_16x16x32_f16(false, a, false, b, (short)0, c,
                                                false, false);
}

// A fragment: 16x32 f16 tile, row-major, leading dim ld.
// Lane L: row (L&15); K chunks [kb,kb+8) and [kb+16,kb+24), kb=(L>>4)*8.
__device__ __forceinline__ v16h ldA(const _Float16* base, int ld) {
  const int lane = threadIdx.x & 31;
  const _Float16* p = base + (size_t)(lane & 15) * ld + ((lane >> 4) << 3);
  v16h r;
  ((v8h*)&r)[0] = *(const v8h*)(p);
  ((v8h*)&r)[1] = *(const v8h*)(p + 16);
  return r;
}

// B fragment: 32x16 col-major B == row-major W[N][K] source.
// Lane L: column (L&15); K chunks [kb,kb+8) and [kb+16,kb+24), kb=(L>>4)*8.
__device__ __forceinline__ v16h ldB(const _Float16* base, int ld) {
  const int lane = threadIdx.x & 31;
  const _Float16* p = base + (size_t)(lane & 15) * ld + ((lane >> 4) << 3);
  v16h r;
  ((v8h*)&r)[0] = *(const v8h*)(p);
  ((v8h*)&r)[1] = *(const v8h*)(p + 16);
  return r;
}

__device__ __forceinline__ void wait_tensor0() {
#if __has_builtin(__builtin_amdgcn_s_wait_tensorcnt)
  __builtin_amdgcn_s_wait_tensorcnt((short)0);
#else
  asm volatile("s_wait_tensorcnt 0x0" ::: "memory");
#endif
}

// TDM 2D tile load: rows x 32 f16, contiguous K (dim0), row stride ldg
// elements. D# per CDNA5 ISA 8.3/8.4: group0 = {count=1, lds_addr,
// global_addr[56:0], type=2}; group1 = {data_size=2B, tensor dims, tile dims,
// dim0 stride}. Generic->LDS byte offset is addr[31:0] (aperture rule).
__device__ __forceinline__ void tdm_load_tile(const _Float16* gsrc,
                                              void* lds_dst, int rows,
                                              int ldg) {
#ifdef HAVE_TDM
  const unsigned long long ga = (unsigned long long)gsrc;
  const unsigned lds_off = (unsigned)(unsigned long long)lds_dst;
  u32x4 g0;
  g0[0] = 1u;                                    // count=1, user mode
  g0[1] = lds_off;                               // lds_addr
  g0[2] = (unsigned)ga;                          // global_addr[31:0]
  g0[3] = (unsigned)((ga >> 32) & 0x01ffffffu) | (2u << 30);  // [56:32],type=2
  i32x8 g1;
  g1[0] = 1 << 16;                               // data_size = 2 bytes
  g1[1] = (int)(32u << 16);                      // tensor_dim0[15:0] = 32
  g1[2] = (int)(((unsigned)rows & 0xffffu) << 16);  // tensor_dim1[15:0]
  g1[3] = (int)(32u << 16);                      // tile_dim0 = 32
  g1[4] = rows;                                  // tile_dim1, tile_dim2=0
  g1[5] = ldg;                                   // tensor_dim0_stride[31:0]
  g1[6] = 0;
  g1[7] = 0;
  i32x4 z4 = {};
#if __clang_major__ >= 23
  i32x8 z8 = {};
  __builtin_amdgcn_tensor_load_to_lds(g0, g1, z4, z4, z8, 0);
#else
  __builtin_amdgcn_tensor_load_to_lds(g0, g1, z4, z4, 0);
#endif
#else
  (void)gsrc; (void)lds_dst; (void)rows; (void)ldg;
#endif
}

__global__ void cvt_f32_f16(const float* __restrict__ in,
                            _Float16* __restrict__ out, int n) {
  int i = blockIdx.x * blockDim.x + threadIdx.x;
  if (i < n) out[i] = (_Float16)in[i];
}

// D = A(MxK f16) @ W.T (W: NxK f16 row-major) + bias.
// 256 thr = 8 waves, block tile 64x256, wave tile 32x64.
// A/B k-slabs staged to LDS by the Tensor Data Mover, double buffered.
constexpr int GM_NORMAL = 0, GM_RELU = 1, GM_TRANS = 2, GM_CUM = 3;

template <int MODE>
__global__ void gemm_bias(const _Float16* __restrict__ A, int lda,
                          const _Float16* __restrict__ W, int ldw,
                          const float* __restrict__ bias,
                          _Float16* __restrict__ out, int ldo,
                          float* __restrict__ out32,
                          int M, int N, int K) {
  __shared__ _Float16 As[2][64 * 32];    // 8 KB
  __shared__ _Float16 Bs[2][256 * 32];   // 32 KB

  const int tid = threadIdx.x;
  const int wave = tid >> 5;
  const int lane = tid & 31;
  const int mb = blockIdx.x * 64;
  const int nb = blockIdx.y * 256;
  const int wm = (wave & 1) * 32;   // wave M offset within block tile
  const int wn = (wave >> 1) * 64;  // wave N offset within block tile

  auto stage = [&](int buf, int k0) {
#ifdef HAVE_TDM
    if (wave == 0) {
      tdm_load_tile(A + (size_t)mb * lda + k0, &As[buf][0], 64, lda);
      tdm_load_tile(W + (size_t)nb * ldw + k0, &Bs[buf][0], 256, ldw);
    }
#else
    {  // cooperative copy fallback: 8-half vectors
      int ch = tid;  // A: 64*32/8 = 256 chunks
      int r = ch >> 2, kc = (ch & 3) * 8;
      *(v8h*)&As[buf][r * 32 + kc] =
          *(const v8h*)&A[(size_t)(mb + r) * lda + k0 + kc];
#pragma unroll
      for (int i = 0; i < 4; ++i) {  // B: 1024 chunks
        int cb = tid + i * 256;
        int rb = cb >> 2, kb = (cb & 3) * 8;
        *(v8h*)&Bs[buf][rb * 32 + kb] =
            *(const v8h*)&W[(size_t)(nb + rb) * ldw + k0 + kb];
      }
    }
#endif
  };

  v8f c[2][4] = {};
  stage(0, 0);
  int cur = 0;
  for (int k0 = 0; k0 < K; k0 += 32) {
#ifdef HAVE_TDM
    if (wave == 0) wait_tensor0();
#endif
    __syncthreads();  // buf[cur] visible to all waves
    if (k0 + 32 < K) stage(cur ^ 1, k0 + 32);

    v16h a0 = ldA(&As[cur][wm * 32], 32);
    v16h a1 = ldA(&As[cur][(wm + 16) * 32], 32);
#pragma unroll
    for (int t = 0; t < 4; ++t) {
      v16h b = ldB(&Bs[cur][(wn + t * 16) * 32], 32);
      c[0][t] = wmma32(a0, b, c[0][t]);
      c[1][t] = wmma32(a1, b, c[1][t]);
    }
    __syncthreads();  // reads of buf[cur] done before it is refilled
    cur ^= 1;
  }

  const int cm = (lane >> 4) << 3;
  const int cn = lane & 15;
  for (int im = 0; im < 2; ++im)
    for (int t = 0; t < 4; ++t) {
      const int colg = nb + wn + t * 16 + cn;
      const float bv = bias[colg];
#pragma unroll
      for (int j = 0; j < 8; ++j) {
        const int rowg = mb + wm + im * 16 + cm + j;
        float v = c[im][t][j] + bv;
        if (MODE == GM_RELU) v = v > 0.f ? v : 0.f;
        if (MODE == GM_TRANS) {
          out[(size_t)colg * ldo + rowg] = (_Float16)v;
        } else {
          out[(size_t)rowg * ldo + colg] = (_Float16)v;
          if (MODE == GM_CUM) out32[(size_t)rowg * ldo + colg] = v;
        }
      }
    }
}

// Attention: one block = (16 query rows) x (1 head) x (1 window).
// Q,K row-major [rows][EMB] (head slice via +h*DH); Vt transposed [feat][key].
__global__ void attn_kernel(const _Float16* __restrict__ Q, long long qws,
                            const _Float16* __restrict__ Km, long long kws,
                            const _Float16* __restrict__ Vt, int vt_ld, int vws,
                            _Float16* __restrict__ Out, long long ows,
                            float scale) {
  __shared__ _Float16 Pl[16 * 256];
  __shared__ float red[8 * 16];
  __shared__ float rowstat[16];
  __shared__ float Opart[16 * 64];

  const int tid = threadIdx.x;
  const int wave = tid >> 5;
  const int lane = tid & 31;
  const int half = lane >> 4;
  const int l15 = lane & 15;
  const int qb = blockIdx.x;
  const int h = blockIdx.y;
  const long long w = blockIdx.z;

  const _Float16* Qw = Q + w * qws + (size_t)qb * 16 * EMB + h * DH;
  const _Float16* Kw = Km + w * kws + h * DH;
  const _Float16* Vw = Vt + (size_t)h * DH * vt_ld + w * (long long)vws;
  _Float16* Ow = Out + w * ows + (size_t)qb * 16 * EMB + h * DH;

  // ---- S = Q @ K^T (this wave: key tiles wave*32 .. +32) ----
  v8f s0 = {}, s1 = {};
  for (int kk = 0; kk < DH; kk += 32) {
    v16h a = ldA(Qw + kk, EMB);
    v16h b0 = ldB(Kw + (size_t)(wave * 32) * EMB + kk, EMB);
    v16h b1 = ldB(Kw + (size_t)(wave * 32 + 16) * EMB + kk, EMB);
    s0 = wmma32(a, b0, s0);
    s1 = wmma32(a, b1, s1);
  }
#pragma unroll
  for (int j = 0; j < 8; ++j) { s0[j] *= scale; s1[j] *= scale; }

  // ---- row max across 256 keys ----
#pragma unroll
  for (int j = 0; j < 8; ++j) {
    float m = fmaxf(s0[j], s1[j]);
    for (int o = 8; o >= 1; o >>= 1) m = fmaxf(m, __shfl_xor(m, o));
    if (l15 == j) red[wave * 16 + half * 8 + j] = m;
  }
  __syncthreads();
  if (tid < 16) {
    float m = -3.4e38f;
    for (int wv = 0; wv < 8; ++wv) m = fmaxf(m, red[wv * 16 + tid]);
    rowstat[tid] = m;
  }
  __syncthreads();

  // ---- exp, stage P to LDS (f16), row sums ----
#pragma unroll
  for (int j = 0; j < 8; ++j) {
    const int r = half * 8 + j;
    const float m = rowstat[r];
    const float e0 = __expf(s0[j] - m);
    const float e1 = __expf(s1[j] - m);
    Pl[r * 256 + wave * 32 + l15] = (_Float16)e0;
    Pl[r * 256 + wave * 32 + 16 + l15] = (_Float16)e1;
    float sm = e0 + e1;
    for (int o = 8; o >= 1; o >>= 1) sm += __shfl_xor(sm, o);
    if (l15 == j) red[wave * 16 + r] = sm;
  }
  __syncthreads();
  if (tid < 16) {
    float sm = 0.f;
    for (int wv = 0; wv < 8; ++wv) sm += red[wv * 16 + tid];
    rowstat[tid] = sm;
  }
  __syncthreads();

  // ---- O = P @ V : wave&3 -> dh tile, wave>>2 -> key half ----
  const int nt = wave & 3;
  const int kh = wave >> 2;
  v8f o = {};
  for (int kk = kh * 128; kk < kh * 128 + 128; kk += 32) {
    v16h a = ldA(&Pl[kk], 256);
    v16h b = ldB(Vw + (size_t)(nt * 16) * vt_ld + kk, vt_ld);
    o = wmma32(a, b, o);
  }
  if (kh == 1) {
#pragma unroll
    for (int j = 0; j < 8; ++j)
      Opart[(half * 8 + j) * 64 + nt * 16 + l15] = o[j];
  }
  __syncthreads();
  if (kh == 0) {
#pragma unroll
    for (int j = 0; j < 8; ++j) {
      const int r = half * 8 + j;
      const float v = (o[j] + Opart[r * 64 + nt * 16 + l15]) / rowstat[r];
      Ow[(size_t)r * EMB + nt * 16 + l15] = (_Float16)v;
    }
  }
}

extern "C" void kernel_launch(void* const* d_in, const int* in_sizes, int n_in,
                              void* d_out, int out_size, void* d_ws,
                              size_t ws_size, hipStream_t stream) {
  (void)in_sizes; (void)n_in; (void)out_size; (void)ws_size;
  const float* path   = (const float*)d_in[0];
  const float* w_in1  = (const float*)d_in[1];
  const float* b_in1  = (const float*)d_in[2];
  const float* w_out1 = (const float*)d_in[3];
  const float* b_out1 = (const float*)d_in[4];
  const float* w_lin  = (const float*)d_in[5];
  const float* b_lin  = (const float*)d_in[6];
  const float* w_in2  = (const float*)d_in[7];
  const float* b_in2  = (const float*)d_in[8];
  const float* w_out2 = (const float*)d_in[9];
  const float* b_out2 = (const float*)d_in[10];
  float* out = (float*)d_out;

  size_t off = 0;
  _Float16* base = (_Float16*)d_ws;
  auto take = [&](size_t n) { _Float16* p = base + off; off += n; return p; };
  _Float16* wf_in1  = take((size_t)3 * EMB * EMB);
  _Float16* wf_out1 = take((size_t)EMB * EMB);
  _Float16* wf_lin  = take((size_t)EMB * EMB);
  _Float16* wf_in2  = take((size_t)3 * EMB * EMB);
  _Float16* wf_out2 = take((size_t)EMB * EMB);
  _Float16* pathh   = take((size_t)SEQ_LEN * EMB);
  _Float16* Q1      = take((size_t)SEQ_LEN * EMB);
  _Float16* K1      = take((size_t)SEQ_LEN * EMB);
  _Float16* V1t     = take((size_t)EMB * SEQ_LEN);
  _Float16* bufA    = take((size_t)NWIN * KWIN * EMB);   // attn1 -> X
  _Float16* bufB    = take((size_t)NWIN * KWIN * EMB);   // proj1 -> Q2(all)
  _Float16* cumh    = take((size_t)KWIN * EMB);
  _Float16* K2b     = take((size_t)KWIN * EMB);
  _Float16* V2t     = take((size_t)EMB * KWIN);
  _Float16* at2     = take((size_t)KWIN * EMB);
  float* cum32      = (float*)(base + off);

  const float scale = 0.125f;  // 1/sqrt(64)

  auto cvt = [&](const float* s, _Float16* d, int n) {
    cvt_f32_f16<<<dim3((n + 255) / 256), 256, 0, stream>>>(s, d, n);
  };
  cvt(path,   pathh,   SEQ_LEN * EMB);
  cvt(w_in1,  wf_in1,  3 * EMB * EMB);
  cvt(w_out1, wf_out1, EMB * EMB);
  cvt(w_lin,  wf_lin,  EMB * EMB);
  cvt(w_in2,  wf_in2,  3 * EMB * EMB);
  cvt(w_out2, wf_out2, EMB * EMB);

  // QKV projection of path (shared by all overlapping windows).
  gemm_bias<GM_NORMAL><<<dim3(SEQ_LEN / 64, EMB / 256), 256, 0, stream>>>(
      pathh, EMB, wf_in1, EMB, b_in1, Q1, EMB, nullptr, SEQ_LEN, EMB, EMB);
  gemm_bias<GM_NORMAL><<<dim3(SEQ_LEN / 64, EMB / 256), 256, 0, stream>>>(
      pathh, EMB, wf_in1 + (size_t)EMB * EMB, EMB, b_in1 + EMB, K1, EMB,
      nullptr, SEQ_LEN, EMB, EMB);
  gemm_bias<GM_TRANS><<<dim3(SEQ_LEN / 64, EMB / 256), 256, 0, stream>>>(
      pathh, EMB, wf_in1 + (size_t)2 * EMB * EMB, EMB, b_in1 + 2 * EMB, V1t,
      SEQ_LEN, nullptr, SEQ_LEN, EMB, EMB);

  // Batched sliding-window attention (window w = rows w*16..+255).
  attn_kernel<<<dim3(KWIN / 16, NHEADS, NWIN), 256, 0, stream>>>(
      Q1, 16LL * EMB, K1, 16LL * EMB, V1t, SEQ_LEN, 16, bufA,
      (long long)KWIN * EMB, scale);

  const int MW = NWIN * KWIN;  // 61440
  // out-proj 1
  gemm_bias<GM_NORMAL><<<dim3(MW / 64, EMB / 256), 256, 0, stream>>>(
      bufA, EMB, wf_out1, EMB, b_out1, bufB, EMB, nullptr, MW, EMB, EMB);
  // linear + relu -> X (reuse bufA)
  gemm_bias<GM_RELU><<<dim3(MW / 64, EMB / 256), 256, 0, stream>>>(
      bufB, EMB, wf_lin, EMB, b_lin, bufA, EMB, nullptr, MW, EMB, EMB);
  // Q-projection of all windows for the scan (hoisted out of serial loop)
  gemm_bias<GM_NORMAL><<<dim3(MW / 64, EMB / 256), 256, 0, stream>>>(
      bufA, EMB, wf_in2, EMB, b_in2, bufB, EMB, nullptr, MW, EMB, EMB);

  // cum_att = x[0]
  hipMemcpyAsync(cumh, bufA, (size_t)KWIN * EMB * sizeof(_Float16),
                 hipMemcpyDeviceToDevice, stream);

  // Sequential recurrent attention accumulation (239 dependent steps).
  for (int i = 1; i < NWIN; ++i) {
    gemm_bias<GM_NORMAL><<<dim3(KWIN / 64, EMB / 256), 256, 0, stream>>>(
        cumh, EMB, wf_in2 + (size_t)EMB * EMB, EMB, b_in2 + EMB, K2b, EMB,
        nullptr, KWIN, EMB, EMB);
    gemm_bias<GM_TRANS><<<dim3(KWIN / 64, EMB / 256), 256, 0, stream>>>(
        cumh, EMB, wf_in2 + (size_t)2 * EMB * EMB, EMB, b_in2 + 2 * EMB, V2t,
        KWIN, nullptr, KWIN, EMB, EMB);
    attn_kernel<<<dim3(KWIN / 16, NHEADS, 1), 256, 0, stream>>>(
        bufB + (size_t)i * KWIN * EMB, 0, K2b, 0, V2t, KWIN, 0, at2, 0, scale);
    gemm_bias<GM_CUM><<<dim3(KWIN / 64, EMB / 256), 256, 0, stream>>>(
        at2, EMB, wf_out2, EMB, b_out2, cumh, EMB, cum32, KWIN, EMB, EMB);
  }

  hipMemcpyAsync(out, cum32, (size_t)KWIN * EMB * sizeof(float),
                 hipMemcpyDeviceToDevice, stream);
}
